// MPNNet_21345987461256
// MI455X (gfx1250) — matrix operand order repaired and checked by reading, not compiled, not run on hardware
//
#include <hip/hip_runtime.h>
#include <cstdint>
#include <cstddef>

#define N_NODES 30000
#define N_EDGES 60000
#define N_MOL   1200
#define DIM     64
#define NF      17
#define EF      4
#define HID     128
#define GD      192   /* 3*DIM GRU gates */
#define LG      256   /* 4*DIM LSTM gates */
#define NSTEPS  12
#define NS2S    3
#define MSG_CHUNK 8   /* edges per block in TDM message kernel */

typedef __attribute__((ext_vector_type(16))) _Float16 v16h;
typedef __attribute__((ext_vector_type(8)))  float    v8f;
typedef __attribute__((ext_vector_type(4)))  unsigned u32x4;
typedef __attribute__((ext_vector_type(8)))  int      i32x8;
typedef __attribute__((ext_vector_type(4)))  int      i32x4;

__device__ __forceinline__ float leakyf(float v) { return v >= 0.f ? v : 0.01f * v; }
__device__ __forceinline__ float sigm(float v)   { return 1.f / (1.f + expf(-v)); }

// monotonic unsigned key for float atomicMax
__device__ __forceinline__ unsigned fkey(float f) {
  unsigned u = __float_as_uint(f);
  return (u & 0x80000000u) ? ~u : (u | 0x80000000u);
}
__device__ __forceinline__ float fdec(unsigned k) {
  return (k & 0x80000000u) ? __uint_as_float(k ^ 0x80000000u) : __uint_as_float(~k);
}

// ---------------------------------------------------------------------------
// WMMA helpers (CDNA5 v_wmma_f32_16x16x32_f16, wave32) contiguous-K order in A and B
// A fragment: 16 contiguous K per lane (K in kb..kb+15, kb = k0 + 16*(lane/16))
// B fragments are PRE-PACKED fragment-major: one contiguous v16h per lane.
// ---------------------------------------------------------------------------
__device__ __forceinline__ v16h load_a_frag(const _Float16* __restrict__ A, int lda,
                                            int m0, int k0, int lane) {
  int row = m0 + (lane & 15);
  int kb  = k0 + ((lane & 16) ? 16 : 0);
  const _Float16* p = A + (size_t)row * lda + kb;
  v16h a;
#pragma unroll
  for (int j = 0; j < 8; ++j)  a[j] = p[j];        // K = kb+0..7
#pragma unroll
  for (int j = 8; j < 16; ++j) a[j] = p[j];        // K = kb+8..15
  return a;
}

__device__ __forceinline__ v8f wmma_step(v16h a, v16h b, v8f c) {
  return __builtin_amdgcn_wmma_f32_16x16x32_f16(false, a, false, b, (short)0, c,
                                                false, false);
}

// ---------------------------------------------------------------------------
// Utility kernels
// ---------------------------------------------------------------------------
__global__ void k_fill_f32(float* p, float v, int n) {
  for (int i = blockIdx.x * blockDim.x + threadIdx.x; i < n; i += gridDim.x * blockDim.x)
    p[i] = v;
}
__global__ void k_fill_u32(unsigned* p, unsigned v, int n) {
  for (int i = blockIdx.x * blockDim.x + threadIdx.x; i < n; i += gridDim.x * blockDim.x)
    p[i] = v;
}

// pack row-major B [K, Ncols] f32 -> fragment-major f16:
// dst[((nt*KT + kt)*32 + lane)*16 + j] = B[kt*32 + ((lane&16)?16:0) + j][nt*16 + (lane&15)]
__global__ void k_pack_b(const float* __restrict__ src, _Float16* __restrict__ dst,
                         int K, int Ncols) {
  int KT = K / 32;
  int total = (Ncols / 16) * KT * 32 * 16;
  for (int idx = blockIdx.x * blockDim.x + threadIdx.x; idx < total;
       idx += gridDim.x * blockDim.x) {
    int j    = idx & 15;
    int lane = (idx >> 4) & 31;
    int kt   = (idx >> 9) % KT;
    int nt   = idx / (KT << 9);
    int col  = nt * 16 + (lane & 15);
    int k    = kt * 32 + ((lane & 16) ? 16 : 0) + j;
    dst[idx] = (_Float16)src[(size_t)k * Ncols + col];
  }
}
// same, but source is [Ncols, K] row-major (torch weight [out,in]); B[k][n]=src[n][k]
__global__ void k_pack_bt(const float* __restrict__ src, _Float16* __restrict__ dst,
                          int K, int Ncols) {
  int KT = K / 32;
  int total = (Ncols / 16) * KT * 32 * 16;
  for (int idx = blockIdx.x * blockDim.x + threadIdx.x; idx < total;
       idx += gridDim.x * blockDim.x) {
    int j    = idx & 15;
    int lane = (idx >> 4) & 31;
    int kt   = (idx >> 9) % KT;
    int nt   = idx / (KT << 9);
    int col  = nt * 16 + (lane & 15);
    int k    = kt * 32 + ((lane & 16) ? 16 : 0) + j;
    dst[idx] = (_Float16)src[(size_t)col * K + k];
  }
}

// ---------------------------------------------------------------------------
// lin0: h = leaky(x @ lin0_w + b)   [N,17]@[17,64]
// ---------------------------------------------------------------------------
__global__ void k_lin0(const float* __restrict__ x, const float* __restrict__ w,
                       const float* __restrict__ b, float* __restrict__ h32,
                       _Float16* __restrict__ h16) {
  int idx = blockIdx.x * blockDim.x + threadIdx.x;
  if (idx >= N_NODES * DIM) return;
  int n = idx / DIM, j = idx % DIM;
  float acc = b[j];
#pragma unroll
  for (int f = 0; f < NF; ++f) acc += x[(size_t)n * NF + f] * w[(size_t)f * DIM + j];
  acc = leakyf(acc);
  h32[idx] = acc;
  h16[idx] = (_Float16)acc;
}

// edge hidden = leaky(edge_attr @ en_w1 + en_b1)   [E,4]@[4,128] -> f16
__global__ void k_edge_hidden(const float* __restrict__ ea, const float* __restrict__ w,
                              const float* __restrict__ b, _Float16* __restrict__ eh) {
  int idx = blockIdx.x * blockDim.x + threadIdx.x;
  if (idx >= N_EDGES * HID) return;
  int e = idx / HID, hcol = idx % HID;
  float acc = b[hcol];
#pragma unroll
  for (int f = 0; f < EF; ++f) acc += ea[(size_t)e * EF + f] * w[(size_t)f * HID + hcol];
  eh[idx] = (_Float16)leakyf(acc);
}

// ---------------------------------------------------------------------------
// WMMA GEMMs, 16x32 output per wave (two N-tiles share one A fragment)
// ---------------------------------------------------------------------------
__global__ void k_gemm_bias_h(const _Float16* __restrict__ A, const _Float16* __restrict__ Bp,
                              const float* __restrict__ bias, _Float16* __restrict__ D,
                              int lda, int K, int ldd) {
  int m0 = blockIdx.x * 16, nt0 = blockIdx.y * 2, lane = threadIdx.x;
  int KT = K / 32;
  const v16h* Bv = (const v16h*)Bp;
  v8f acc0 = {}, acc1 = {};
  for (int kt = 0; kt < KT; ++kt) {
    v16h a  = load_a_frag(A, lda, m0, kt * 32, lane);
    v16h b0 = Bv[((size_t)nt0 * KT + kt) * 32 + lane];
    v16h b1 = Bv[((size_t)(nt0 + 1) * KT + kt) * 32 + lane];
    acc0 = wmma_step(a, b0, acc0);
    acc1 = wmma_step(a, b1, acc1);
  }
  int rbase = m0 + ((lane & 16) ? 8 : 0);
  int c0 = nt0 * 16 + (lane & 15), c1 = c0 + 16;
  float bv0 = bias[c0], bv1 = bias[c1];
#pragma unroll
  for (int r = 0; r < 8; ++r) {
    D[(size_t)(rbase + r) * ldd + c0] = (_Float16)(acc0[r] + bv0);
    D[(size_t)(rbase + r) * ldd + c1] = (_Float16)(acc1[r] + bv1);
  }
}

__global__ void k_gemm_bias_f(const _Float16* __restrict__ A, const _Float16* __restrict__ Bp,
                              const float* __restrict__ bias, float* __restrict__ D,
                              int lda, int K, int ldd) {
  int m0 = blockIdx.x * 16, nt0 = blockIdx.y * 2, lane = threadIdx.x;
  int KT = K / 32;
  const v16h* Bv = (const v16h*)Bp;
  v8f acc0 = {}, acc1 = {};
  for (int kt = 0; kt < KT; ++kt) {
    v16h a  = load_a_frag(A, lda, m0, kt * 32, lane);
    v16h b0 = Bv[((size_t)nt0 * KT + kt) * 32 + lane];
    v16h b1 = Bv[((size_t)(nt0 + 1) * KT + kt) * 32 + lane];
    acc0 = wmma_step(a, b0, acc0);
    acc1 = wmma_step(a, b1, acc1);
  }
  int rbase = m0 + ((lane & 16) ? 8 : 0);
  int c0 = nt0 * 16 + (lane & 15), c1 = c0 + 16;
  float bv0 = bias[c0], bv1 = bias[c1];
#pragma unroll
  for (int r = 0; r < 8; ++r) {
    D[(size_t)(rbase + r) * ldd + c0] = acc0[r] + bv0;
    D[(size_t)(rbase + r) * ldd + c1] = acc1[r] + bv1;
  }
}

// WMMA + NNConv epilogue: m = leaky(h@root_w + agg/deg + conv_b) -> f16
__global__ void k_gemm_m(const _Float16* __restrict__ A, const _Float16* __restrict__ Bp,
                         const float* __restrict__ agg, const float* __restrict__ deg,
                         const float* __restrict__ convb, _Float16* __restrict__ M) {
  int m0 = blockIdx.x * 16, nt0 = blockIdx.y * 2, lane = threadIdx.x;
  const int KT = DIM / 32;
  const v16h* Bv = (const v16h*)Bp;
  v8f acc0 = {}, acc1 = {};
#pragma unroll
  for (int kt = 0; kt < KT; ++kt) {
    v16h a  = load_a_frag(A, DIM, m0, kt * 32, lane);
    v16h b0 = Bv[((size_t)nt0 * KT + kt) * 32 + lane];
    v16h b1 = Bv[((size_t)(nt0 + 1) * KT + kt) * 32 + lane];
    acc0 = wmma_step(a, b0, acc0);
    acc1 = wmma_step(a, b1, acc1);
  }
  int rbase = m0 + ((lane & 16) ? 8 : 0);
  int c0 = nt0 * 16 + (lane & 15), c1 = c0 + 16;
  float bv0 = convb[c0], bv1 = convb[c1];
#pragma unroll
  for (int r = 0; r < 8; ++r) {
    int row = rbase + r;
    float dg = fmaxf(deg[row], 1.f);
    float v0 = acc0[r] + agg[(size_t)row * DIM + c0] / dg + bv0;
    float v1 = acc1[r] + agg[(size_t)row * DIM + c1] / dg + bv1;
    M[(size_t)row * DIM + c0] = (_Float16)leakyf(v0);
    M[(size_t)row * DIM + c1] = (_Float16)leakyf(v1);
  }
}

// ---------------------------------------------------------------------------
// degree (scatter count)
// ---------------------------------------------------------------------------
__global__ void k_deg(const int* __restrict__ dst, float* __restrict__ deg) {
  int e = blockIdx.x * blockDim.x + threadIdx.x;
  if (e < N_EDGES) atomicAdd(&deg[dst[e]], 1.f);
}

// ---------------------------------------------------------------------------
// per-edge message msg = h[src] @ W_e (f16 64x64), scatter-add into agg.
// One wave per block, MSG_CHUNK edges per block. Each 8KB W tile is DMA'd
// into LDS by the Tensor Data Mover (double-buffered), synchronized with
// s_wait_tensorcnt; compute reads LDS, scatters with global f32 atomics.
// ---------------------------------------------------------------------------
__global__ void k_msg(const float* __restrict__ h32, const _Float16* __restrict__ we,
                      const int* __restrict__ src, const int* __restrict__ dst,
                      float* __restrict__ agg) {
  __shared__ _Float16 smem[2][DIM * DIM];
  int lane  = threadIdx.x;               // 0..31
  int ebase = blockIdx.x * MSG_CHUNK;
  int c0 = lane * 2;

#if defined(__AMDGCN__) && __has_builtin(__builtin_amdgcn_tensor_load_to_lds)
  unsigned ldsa[2] = { (unsigned)(size_t)(&smem[0][0]),
                       (unsigned)(size_t)(&smem[1][0]) };
  auto issue = [&](int t) {
    unsigned long long ga =
        (unsigned long long)(size_t)(we + (size_t)(ebase + t) * (DIM * DIM));
    u32x4 g0 = (u32x4)0;
    g0[0] = 1u;                                        // count=1, user mode
    g0[1] = ldsa[t & 1];                               // lds_addr
    g0[2] = (unsigned)(ga & 0xFFFFFFFFu);              // global_addr[31:0]
    g0[3] = (unsigned)((ga >> 32) & 0x01FFFFFFu) | 0x80000000u;  // addr[56:32]|type=2
    i32x8 g1 = (i32x8)0;
    g1[0] = 0x30000;                                   // data_size = 8B units
    g1[1] = (int)(1024u << 16);                        // tensor_dim0 = 1024
    g1[2] = (int)(1u << 16);                           // tensor_dim1 = 1
    g1[3] = (int)(1024u << 16);                        // tile_dim0 = 1024
    g1[5] = 1024;                                      // tensor_dim0_stride
    i32x4 gz = (i32x4)0;
    asm volatile("s_wait_dscnt 0" ::: "memory");       // buffer reuse guard
#if __clang_major__ <= 22
    __builtin_amdgcn_tensor_load_to_lds(g0, g1, gz, gz, 0);
#else
    i32x8 gz8 = (i32x8)0;
    __builtin_amdgcn_tensor_load_to_lds(g0, g1, gz, gz, gz8, 0);
#endif
  };

  issue(0);
  for (int t = 0; t < MSG_CHUNK; ++t) {
    if (t + 1 < MSG_CHUNK) {
      issue(t + 1);
      __builtin_amdgcn_s_wait_tensorcnt(1);            // tile t resident
    } else {
      __builtin_amdgcn_s_wait_tensorcnt(0);
    }
    int e = ebase + t;
    const _Float16* W = &smem[t & 1][0];
    const float* s = h32 + (size_t)src[e] * DIM;
    float a0 = 0.f, a1 = 0.f;
#pragma unroll 8
    for (int d = 0; d < DIM; ++d) {
      float sv = s[d];
      a0 += sv * (float)W[d * DIM + c0];
      a1 += sv * (float)W[d * DIM + c0 + 1];
    }
    int dd = dst[e];
    atomicAdd(&agg[(size_t)dd * DIM + c0],     a0);
    atomicAdd(&agg[(size_t)dd * DIM + c0 + 1], a1);
  }
#else
  // fallback: direct global loads
  for (int t = 0; t < MSG_CHUNK; ++t) {
    int e = ebase + t;
    const _Float16* W = we + (size_t)e * (DIM * DIM);
    const float* s = h32 + (size_t)src[e] * DIM;
    float a0 = 0.f, a1 = 0.f;
    for (int d = 0; d < DIM; ++d) {
      float sv = s[d];
      a0 += sv * (float)W[d * DIM + c0];
      a1 += sv * (float)W[d * DIM + c0 + 1];
    }
    int dd = dst[e];
    atomicAdd(&agg[(size_t)dd * DIM + c0],     a0);
    atomicAdd(&agg[(size_t)dd * DIM + c0 + 1], a1);
  }
#endif
}

// ---------------------------------------------------------------------------
// GRU elementwise (gate order r,z,n)
// ---------------------------------------------------------------------------
__global__ void k_gru(const float* __restrict__ gi, const float* __restrict__ gh,
                      float* __restrict__ h32, _Float16* __restrict__ h16) {
  int idx = blockIdx.x * blockDim.x + threadIdx.x;
  if (idx >= N_NODES * DIM) return;
  int n = idx / DIM, d = idx % DIM;
  size_t base = (size_t)n * GD;
  float r  = sigm(gi[base + d]       + gh[base + d]);
  float z  = sigm(gi[base + DIM + d] + gh[base + DIM + d]);
  float nn = tanhf(gi[base + 2 * DIM + d] + r * gh[base + 2 * DIM + d]);
  float hv = (1.f - z) * nn + z * h32[idx];
  h32[idx] = hv;
  h16[idx] = (_Float16)hv;
}

// ---------------------------------------------------------------------------
// Set2Set LSTM step: one block per molecule, 256 threads = 256 gate elems
// ---------------------------------------------------------------------------
__global__ void k_s2s_lstm(const float* __restrict__ qstar, float* __restrict__ hl,
                           float* __restrict__ cl, const float* __restrict__ wih,
                           const float* __restrict__ whh, const float* __restrict__ bih,
                           const float* __restrict__ bhh) {
  int b = blockIdx.x;
  int j = threadIdx.x;                    // 0..255
  float acc = bih[j] + bhh[j];
  for (int k = 0; k < 2 * DIM; ++k)
    acc += qstar[(size_t)b * (2 * DIM) + k] * wih[(size_t)j * (2 * DIM) + k];
  for (int k = 0; k < DIM; ++k)
    acc += hl[(size_t)b * DIM + k] * whh[(size_t)j * DIM + k];
  __shared__ float sg[LG];
  sg[j] = acc;
  __syncthreads();
  if (j < DIM) {
    float gi = sg[j], gf = sg[DIM + j], gg = sg[2 * DIM + j], go = sg[3 * DIM + j];
    float c = sigm(gf) * cl[(size_t)b * DIM + j] + sigm(gi) * tanhf(gg);
    cl[(size_t)b * DIM + j] = c;
    hl[(size_t)b * DIM + j] = sigm(go) * tanhf(c);
  }
}

// attention energies + segment max (keyed atomicMax)
__global__ void k_e_att(const float* __restrict__ h32, const float* __restrict__ hl,
                        const int* __restrict__ batch, float* __restrict__ e_arr,
                        unsigned* __restrict__ emk) {
  int n = blockIdx.x * blockDim.x + threadIdx.x;
  if (n >= N_NODES) return;
  int b = batch[n];
  float acc = 0.f;
#pragma unroll 8
  for (int d = 0; d < DIM; ++d)
    acc += h32[(size_t)n * DIM + d] * hl[(size_t)b * DIM + d];
  e_arr[n] = acc;
  atomicMax(&emk[b], fkey(acc));
}

// unnormalized softmax weights + segment denom
__global__ void k_aun(const float* __restrict__ e_arr, const unsigned* __restrict__ emk,
                      const int* __restrict__ batch, float* __restrict__ aun,
                      float* __restrict__ denom) {
  int n = blockIdx.x * blockDim.x + threadIdx.x;
  if (n >= N_NODES) return;
  int b = batch[n];
  unsigned k = emk[b];
  float emax = (k == 0u) ? 0.f : fdec(k);
  float a = expf(e_arr[n] - emax);
  aun[n] = a;
  atomicAdd(&denom[b], a);
}

// weighted readout r = segment_sum(a * h)
__global__ void k_rsum(const float* __restrict__ aun, const float* __restrict__ denom,
                       const int* __restrict__ batch, const float* __restrict__ h32,
                       float* __restrict__ rsum) {
  int idx = blockIdx.x * blockDim.x + threadIdx.x;
  if (idx >= N_NODES * DIM) return;
  int n = idx / DIM, d = idx % DIM;
  int b = batch[n];
  float a = aun[n] / fmaxf(denom[b], 1e-16f);
  atomicAdd(&rsum[(size_t)b * DIM + d], a * h32[idx]);
}

__global__ void k_qstar(const float* __restrict__ hl, const float* __restrict__ rsum,
                        float* __restrict__ qstar) {
  int idx = blockIdx.x * blockDim.x + threadIdx.x;
  if (idx >= N_MOL * 2 * DIM) return;
  int b = idx / (2 * DIM), j = idx % (2 * DIM);
  qstar[idx] = (j < DIM) ? hl[(size_t)b * DIM + j] : rsum[(size_t)b * DIM + (j - DIM)];
}

__global__ void k_final(const float* __restrict__ qstar, const float* __restrict__ w,
                        const float* __restrict__ b, float* __restrict__ out) {
  int m = blockIdx.x * blockDim.x + threadIdx.x;
  if (m >= N_MOL) return;
  float acc = b[0];
#pragma unroll 8
  for (int j = 0; j < 2 * DIM; ++j) acc += qstar[(size_t)m * (2 * DIM) + j] * w[j];
  out[m] = acc;
}

// ---------------------------------------------------------------------------
// host driver
// ---------------------------------------------------------------------------
extern "C" void kernel_launch(void* const* d_in, const int* in_sizes, int n_in,
                              void* d_out, int out_size, void* d_ws, size_t ws_size,
                              hipStream_t stream) {
  (void)in_sizes; (void)n_in; (void)out_size; (void)ws_size;
  const float* x       = (const float*)d_in[0];
  const float* eattr   = (const float*)d_in[1];
  const int*   eidx    = (const int*)d_in[2];
  const int*   batch   = (const int*)d_in[3];
  const float* lin0_w  = (const float*)d_in[4];
  const float* lin0_b  = (const float*)d_in[5];
  const float* en_w1   = (const float*)d_in[6];
  const float* en_b1   = (const float*)d_in[7];
  const float* en_w2   = (const float*)d_in[8];
  const float* en_b2   = (const float*)d_in[9];
  const float* root_w  = (const float*)d_in[10];
  const float* conv_b  = (const float*)d_in[11];
  const float* gru_wih = (const float*)d_in[12];
  const float* gru_whh = (const float*)d_in[13];
  const float* gru_bih = (const float*)d_in[14];
  const float* gru_bhh = (const float*)d_in[15];
  const float* ls_wih  = (const float*)d_in[16];
  const float* ls_whh  = (const float*)d_in[17];
  const float* ls_bih  = (const float*)d_in[18];
  const float* ls_bhh  = (const float*)d_in[19];
  const float* lin3_w  = (const float*)d_in[20];
  const float* lin3_b  = (const float*)d_in[21];
  const int* srcI = eidx;
  const int* dstI = eidx + N_EDGES;

  char* wp = (char*)d_ws;
  auto carve = [&](size_t bytes) -> void* {
    void* p = (void*)wp;
    wp += (bytes + 255) & ~(size_t)255;
    return p;
  };
  _Float16* we16   = (_Float16*)carve((size_t)N_EDGES * DIM * DIM * 2);
  _Float16* ehid16 = (_Float16*)carve((size_t)N_EDGES * HID * 2);
  float*    h32    = (float*)   carve((size_t)N_NODES * DIM * 4);
  _Float16* h16    = (_Float16*)carve((size_t)N_NODES * DIM * 2);
  _Float16* m16    = (_Float16*)carve((size_t)N_NODES * DIM * 2);
  float*    agg    = (float*)   carve((size_t)N_NODES * DIM * 4);
  float*    gi     = (float*)   carve((size_t)N_NODES * GD * 4);
  float*    gh     = (float*)   carve((size_t)N_NODES * GD * 4);
  float*    deg    = (float*)   carve((size_t)N_NODES * 4);
  _Float16* enw2p  = (_Float16*)carve((size_t)HID * DIM * DIM * 2);   // packed frags
  _Float16* rootp  = (_Float16*)carve((size_t)DIM * DIM * 2);
  _Float16* wihp   = (_Float16*)carve((size_t)DIM * GD * 2);
  _Float16* whhp   = (_Float16*)carve((size_t)DIM * GD * 2);
  float*    e_arr  = (float*)   carve((size_t)N_NODES * 4);
  float*    aun    = (float*)   carve((size_t)N_NODES * 4);
  unsigned* emk    = (unsigned*)carve((size_t)N_MOL * 4);
  float*    denom  = (float*)   carve((size_t)N_MOL * 4);
  float*    rsum   = (float*)   carve((size_t)N_MOL * DIM * 4);
  float*    qstar  = (float*)   carve((size_t)N_MOL * 2 * DIM * 4);
  float*    hl     = (float*)   carve((size_t)N_MOL * DIM * 4);
  float*    cl     = (float*)   carve((size_t)N_MOL * DIM * 4);

  auto blk = [](int n) { return (n + 255) / 256; };

  // --- pack weights into WMMA-fragment-major f16 ---
  k_pack_b <<<blk(HID * DIM * DIM), 256, 0, stream>>>(en_w2, enw2p, HID, DIM * DIM);
  k_pack_b <<<blk(DIM * DIM),       256, 0, stream>>>(root_w, rootp, DIM, DIM);
  k_pack_bt<<<blk(DIM * GD),        256, 0, stream>>>(gru_wih, wihp, DIM, GD);
  k_pack_bt<<<blk(DIM * GD),        256, 0, stream>>>(gru_whh, whhp, DIM, GD);

  // --- degrees ---
  k_fill_f32<<<blk(N_NODES), 256, 0, stream>>>(deg, 0.f, N_NODES);
  k_deg<<<blk(N_EDGES), 256, 0, stream>>>(dstI, deg);

  // --- node embedding ---
  k_lin0<<<blk(N_NODES * DIM), 256, 0, stream>>>(x, lin0_w, lin0_b, h32, h16);

  // --- edge network (loop-invariant, once): WMMA [E,128]@[128,4096] -> f16 ---
  k_edge_hidden<<<blk(N_EDGES * HID), 256, 0, stream>>>(eattr, en_w1, en_b1, ehid16);
  {
    dim3 g(N_EDGES / 16, (DIM * DIM) / 32);
    k_gemm_bias_h<<<g, 32, 0, stream>>>(ehid16, enw2p, en_b2, we16, HID, HID, DIM * DIM);
  }

  // --- message passing steps ---
  for (int s = 0; s < NSTEPS; ++s) {
    k_fill_f32<<<blk(N_NODES * DIM), 256, 0, stream>>>(agg, 0.f, N_NODES * DIM);
    k_msg<<<N_EDGES / MSG_CHUNK, 32, 0, stream>>>(h32, we16, srcI, dstI, agg);
    {
      dim3 g(N_NODES / 16, DIM / 32);
      k_gemm_m<<<g, 32, 0, stream>>>(h16, rootp, agg, deg, conv_b, m16);
    }
    {
      dim3 g(N_NODES / 16, GD / 32);
      k_gemm_bias_f<<<g, 32, 0, stream>>>(m16, wihp, gru_bih, gi, DIM, DIM, GD);
      k_gemm_bias_f<<<g, 32, 0, stream>>>(h16, whhp, gru_bhh, gh, DIM, DIM, GD);
    }
    k_gru<<<blk(N_NODES * DIM), 256, 0, stream>>>(gi, gh, h32, h16);
  }

  // --- Set2Set pooling ---
  k_fill_f32<<<blk(N_MOL * 2 * DIM), 256, 0, stream>>>(qstar, 0.f, N_MOL * 2 * DIM);
  k_fill_f32<<<blk(N_MOL * DIM), 256, 0, stream>>>(hl, 0.f, N_MOL * DIM);
  k_fill_f32<<<blk(N_MOL * DIM), 256, 0, stream>>>(cl, 0.f, N_MOL * DIM);
  for (int it = 0; it < NS2S; ++it) {
    k_s2s_lstm<<<N_MOL, LG, 0, stream>>>(qstar, hl, cl, ls_wih, ls_whh, ls_bih, ls_bhh);
    k_fill_u32<<<blk(N_MOL), 256, 0, stream>>>(emk, 0u, N_MOL);
    k_fill_f32<<<blk(N_MOL), 256, 0, stream>>>(denom, 0.f, N_MOL);
    k_fill_f32<<<blk(N_MOL * DIM), 256, 0, stream>>>(rsum, 0.f, N_MOL * DIM);
    k_e_att<<<blk(N_NODES), 256, 0, stream>>>(h32, hl, batch, e_arr, emk);
    k_aun<<<blk(N_NODES), 256, 0, stream>>>(e_arr, emk, batch, aun, denom);
    k_rsum<<<blk(N_NODES * DIM), 256, 0, stream>>>(aun, denom, batch, h32, rsum);
    k_qstar<<<blk(N_MOL * 2 * DIM), 256, 0, stream>>>(hl, rsum, qstar);
  }

  // --- final linear ---
  k_final<<<blk(N_MOL), 256, 0, stream>>>(qstar, lin3_w, lin3_b, (float*)d_out);
}
